// AGCRNCell_21835613733643
// MI455X (gfx1250) — matrix operand order repaired and checked by reading, not compiled, not run on hardware
//
#include <hip/hip_runtime.h>

// ---------------------------------------------------------------------------
// AGCRN cell for MI455X (gfx1250, wave32): WMMA bf16 hi/lo split precision,
// TDM (tensor_load_to_lds) double-buffered staging in the dominant GEMM.
// ---------------------------------------------------------------------------

typedef __bf16 bvec16 __attribute__((ext_vector_type(16)));
typedef float  fvec8  __attribute__((ext_vector_type(8)));

#define NNODE 2000
#define KPAD  2016              // 2000 padded to multiple of 32
#define BATCH 64
#define DIN   2
#define HDIM  64
#define CIN   66                // DIN + HDIM
#define JTOT  (BATCH * CIN)     // 4224 = 33 * 128
#define KKPAD 160               // 2*CIN = 132 padded to multiple of 32
#define EDIM  10
#define OUTBLK ((size_t)BATCH * NNODE * HDIM)   // 8,192,000

#define TDM_OK __has_builtin(__builtin_amdgcn_tensor_load_to_lds)

static __device__ __forceinline__ float sigmoidf_(float x) {
    return 1.0f / (1.0f + __expf(-x));
}

static __device__ __forceinline__ void split_bf(float v, __bf16& h, __bf16& l) {
    h = (__bf16)v;
    l = (__bf16)(v - (float)h);
}

// A-fragment (16x32 bf16): lane l, row = l&15; elems 0..7 = K at kk0..kk0+7,
// elems 8..15 = K at kk0+16..kk0+23 with kk0 = (l>=16 ? 8 : 0)  -> two b128.
// B-fragment (32x16 bf16): lane l, col = l&15; same K order as A:
// kk0 = (l>=16 ? 8 : 0) -> two b128.
static __device__ __forceinline__ bvec16 ld_frag2(const __bf16* p0, const __bf16* p1) {
    bvec16 f;
    ((uint4*)&f)[0] = *(const uint4*)p0;
    ((uint4*)&f)[1] = *(const uint4*)p1;
    return f;
}

static __device__ __forceinline__ fvec8 wmma_bf16(bvec16 a, bvec16 b, fvec8 c) {
    return __builtin_amdgcn_wmma_f32_16x16x32_bf16(false, a, false, b, (short)0, c,
                                                   false, false);
}

// ---------------------------------------------------------------------------
// TDM: issue one 2D tile load (rows x 32 bf16 elems, row stride `stride_elems`)
// from global into a contiguous LDS region. Rows past `rows` (tensor_dim1)
// read as zero -> the TDM implements the M-edge guard for us.
// D# packing per CDNA5 ISA ch.8 (group0 128b, group1 256b; groups 2/3 zero).
// ---------------------------------------------------------------------------
#if TDM_OK
static __device__ __forceinline__ void tdm_load_tile_2d(unsigned lds_addr,
                                                        const __bf16* gptr,
                                                        unsigned rows,
                                                        unsigned stride_elems) {
    typedef unsigned uvec4 __attribute__((ext_vector_type(4)));
    typedef int ivec8 __attribute__((ext_vector_type(8)));
    typedef int ivec4 __attribute__((ext_vector_type(4)));
    unsigned long long ga = (unsigned long long)(uintptr_t)gptr;
    uvec4 g0;
    g0[0] = 1u;                                     // count=1 (valid user D#)
    g0[1] = lds_addr;                               // lds_addr (bytes)
    g0[2] = (unsigned)ga;                           // global_addr[31:0]
    g0[3] = (unsigned)((ga >> 32) & 0x01FFFFFFu)    // global_addr[56:32]
            | (2u << 30);                           // type=2 ("image")
    ivec8 g1;
    g1[0] = (int)(1u << 16);                        // wg_mask=0, data_size=1 (2B)
    g1[1] = (int)(32u << 16);                       // tensor_dim0=32 (lo16)
    g1[2] = (int)((rows & 0xFFFFu) << 16);          // tdim0 hi | tensor_dim1 lo
    g1[3] = (int)(32u << 16);                       // tdim1 hi | tile_dim0=32
    g1[4] = (int)128;                               // tile_dim1=128, tile_dim2=0
    g1[5] = (int)stride_elems;                      // tensor_dim0_stride lo32
    g1[6] = 0;                                      // stride0 hi | stride1 lo
    g1[7] = 0;                                      // stride1 hi
    ivec4 z4 = {0, 0, 0, 0};
#if defined(__clang_major__) && (__clang_major__ >= 23)
    ivec8 z8 = {0, 0, 0, 0, 0, 0, 0, 0};
    __builtin_amdgcn_tensor_load_to_lds(g0, g1, z4, z4, z8, 0);
#else
    __builtin_amdgcn_tensor_load_to_lds(g0, g1, z4, z4, 0);
#endif
}
#endif

// ---------------------------------------------------------------------------
// K0: supports = softmax(relu(E E^T)) per row -> bf16 hi/lo, K-padded.
// ---------------------------------------------------------------------------
__global__ __launch_bounds__(256) void k0_supports(const float* __restrict__ emb,
                                                   __bf16* __restrict__ sup_hi,
                                                   __bf16* __restrict__ sup_lo) {
    const int r = blockIdx.x;
    const int tid = threadIdx.x;
    __shared__ float g[NNODE];
    __shared__ float red[256];

    float a[EDIM];
#pragma unroll
    for (int d = 0; d < EDIM; ++d) a[d] = emb[r * EDIM + d];

    float lmax = 0.0f;  // relu output >= 0
    for (int m = tid; m < NNODE; m += 256) {
        float dot = 0.0f;
#pragma unroll
        for (int d = 0; d < EDIM; ++d) dot += a[d] * emb[m * EDIM + d];
        dot = fmaxf(dot, 0.0f);
        g[m] = dot;
        lmax = fmaxf(lmax, dot);
    }
    red[tid] = lmax;
    __syncthreads();
    for (int s = 128; s > 0; s >>= 1) {
        if (tid < s) red[tid] = fmaxf(red[tid], red[tid + s]);
        __syncthreads();
    }
    const float rmax = red[0];
    __syncthreads();

    float lsum = 0.0f;
    for (int m = tid; m < NNODE; m += 256) {
        float e = __expf(g[m] - rmax);
        g[m] = e;
        lsum += e;
    }
    red[tid] = lsum;
    __syncthreads();
    for (int s = 128; s > 0; s >>= 1) {
        if (tid < s) red[tid] += red[tid + s];
        __syncthreads();
    }
    const float inv = 1.0f / red[0];

    for (int m = tid; m < NNODE; m += 256) {
        float p = g[m] * inv;
        __bf16 h, l;
        split_bf(p, h, l);
        sup_hi[(size_t)r * KPAD + m] = h;
        sup_lo[(size_t)r * KPAD + m] = l;
    }
    if (tid < KPAD - NNODE) {  // zero K pad
        sup_hi[(size_t)r * KPAD + NNODE + tid] = (__bf16)0.0f;
        sup_lo[(size_t)r * KPAD + NNODE + tid] = (__bf16)0.0f;
    }
}

// ---------------------------------------------------------------------------
// K1: pack concat(x, state)[b,m,i] into
//   Xp[col=b*66+i][k=m]   (column-major B operand, K-padded, bf16 hi/lo)
//   Xg[n=m][b][kk=i]      (identity k=0 slice of stage-3 A operand)
// plus zero the stage-3 K pad region kk in [132,160).
// ---------------------------------------------------------------------------
__global__ __launch_bounds__(256) void k1_pack(const float* __restrict__ xin,
                                               const float* __restrict__ st,
                                               __bf16* __restrict__ Xph,
                                               __bf16* __restrict__ Xpl,
                                               __bf16* __restrict__ Xgh,
                                               __bf16* __restrict__ Xgl) {
    const long T1 = (long)JTOT * KPAD;                        // 8,515,584
    const long T2 = (long)NNODE * BATCH * (KKPAD - 2 * CIN);  // 3,584,000
    long idx = (long)blockIdx.x * 256 + threadIdx.x;
    if (idx < T1) {
        int col = (int)(idx / KPAD);
        int m   = (int)(idx % KPAD);
        int b = col / CIN, i = col % CIN;
        float v = 0.0f;
        if (m < NNODE)
            v = (i < DIN) ? xin[((long)b * NNODE + m) * DIN + i]
                          : st[((long)b * NNODE + m) * HDIM + (i - DIN)];
        __bf16 h, l;
        split_bf(v, h, l);
        Xph[idx] = h;
        Xpl[idx] = l;
        if (m < NNODE) {
            size_t off = (size_t)m * (BATCH * KKPAD) + (size_t)b * KKPAD + i;
            Xgh[off] = h;
            Xgl[off] = l;
        }
    } else if (idx < T1 + T2) {
        long r = idx - T1;
        int n = (int)(r / (BATCH * 28));
        int rem = (int)(r % (BATCH * 28));
        int b = rem / 28;
        int kk = 132 + rem % 28;
        size_t off = (size_t)n * (BATCH * KKPAD) + (size_t)b * KKPAD + kk;
        Xgh[off] = (__bf16)0.0f;
        Xgl[off] = (__bf16)0.0f;
    }
}

// ---------------------------------------------------------------------------
// K2: C[n, j] = sum_k supports[n,k] * Xp[j][k]  via bf16 WMMA hi/lo split.
// Block tile 128x128, 8 waves x (2x4) 16x16 tiles, K step 32.
// TDM pipeline: wave 0 DMAs the next k-step's four tiles into the other LDS
// buffer while all waves run WMMA on the current one (ASYNC via TENSORcnt).
// Epilogue scatters C into Xg[n][b][66+i] (j = b*66+i) as bf16 hi/lo.
// ---------------------------------------------------------------------------
__global__ __launch_bounds__(256) void k2_gemm(const __bf16* __restrict__ Ahi,
                                               const __bf16* __restrict__ Alo,
                                               const __bf16* __restrict__ Bhi,
                                               const __bf16* __restrict__ Blo,
                                               __bf16* __restrict__ Xgh,
                                               __bf16* __restrict__ Xgl) {
    __shared__ __align__(16) __bf16 As_h[2][128 * 32];
    __shared__ __align__(16) __bf16 As_l[2][128 * 32];
    __shared__ __align__(16) __bf16 Bs_h[2][128 * 32];
    __shared__ __align__(16) __bf16 Bs_l[2][128 * 32];

    const int tid = threadIdx.x;
    const int lane = tid & 31;
    const int wv = tid >> 5;
    const int wr = wv & 3;   // 4 row groups of 32
    const int wc = wv >> 2;  // 2 col groups of 64
    const int l15 = lane & 15;
    const int akk = (lane >= 16) ? 8 : 0;
    const int bkk = (lane >= 16) ? 8 : 0;
    const int row0 = blockIdx.x * 128;
    const int col0 = blockIdx.y * 128;
    const int KT = KPAD / 32;  // 63

    int arows = NNODE - row0;
    if (arows > 128) arows = 128;

    fvec8 acc[2][4];
#pragma unroll
    for (int a = 0; a < 2; ++a)
#pragma unroll
        for (int b = 0; b < 4; ++b) {
            fvec8 z = {0.f, 0.f, 0.f, 0.f, 0.f, 0.f, 0.f, 0.f};
            acc[a][b] = z;
        }

#if TDM_OK
    // ---- TDM double-buffered pipeline ----
    if (wv == 0) {  // prologue: DMA k-step 0 into buffer 0
        tdm_load_tile_2d((unsigned)(uintptr_t)&As_h[0][0], Ahi + (size_t)row0 * KPAD,
                         (unsigned)arows, KPAD);
        tdm_load_tile_2d((unsigned)(uintptr_t)&As_l[0][0], Alo + (size_t)row0 * KPAD,
                         (unsigned)arows, KPAD);
        tdm_load_tile_2d((unsigned)(uintptr_t)&Bs_h[0][0], Bhi + (size_t)col0 * KPAD,
                         128u, KPAD);
        tdm_load_tile_2d((unsigned)(uintptr_t)&Bs_l[0][0], Blo + (size_t)col0 * KPAD,
                         128u, KPAD);
    }
    for (int kt = 0; kt < KT; ++kt) {
        const int buf = kt & 1;
        if (wv == 0) __builtin_amdgcn_s_wait_tensorcnt(0);
        __syncthreads();  // buf's tiles visible; all waves done with buf^1
        if (wv == 0 && kt + 1 < KT) {
            const int nb = buf ^ 1;
            const int kg = (kt + 1) * 32;
            tdm_load_tile_2d((unsigned)(uintptr_t)&As_h[nb][0],
                             Ahi + (size_t)row0 * KPAD + kg, (unsigned)arows, KPAD);
            tdm_load_tile_2d((unsigned)(uintptr_t)&As_l[nb][0],
                             Alo + (size_t)row0 * KPAD + kg, (unsigned)arows, KPAD);
            tdm_load_tile_2d((unsigned)(uintptr_t)&Bs_h[nb][0],
                             Bhi + (size_t)col0 * KPAD + kg, 128u, KPAD);
            tdm_load_tile_2d((unsigned)(uintptr_t)&Bs_l[nb][0],
                             Blo + (size_t)col0 * KPAD + kg, 128u, KPAD);
        }
        const __bf16* ash = &As_h[buf][0];
        const __bf16* asl = &As_l[buf][0];
        const __bf16* bsh = &Bs_h[buf][0];
        const __bf16* bsl = &Bs_l[buf][0];

        bvec16 af_h[2], af_l[2];
#pragma unroll
        for (int tr = 0; tr < 2; ++tr) {
            const int arow = wr * 32 + tr * 16 + l15;
            af_h[tr] = ld_frag2(ash + arow * 32 + akk, ash + arow * 32 + akk + 16);
            af_l[tr] = ld_frag2(asl + arow * 32 + akk, asl + arow * 32 + akk + 16);
        }
#pragma unroll
        for (int ct = 0; ct < 4; ++ct) {
            const int bcol = wc * 64 + ct * 16 + l15;
            bvec16 bh = ld_frag2(bsh + bcol * 32 + bkk, bsh + bcol * 32 + bkk + 16);
            bvec16 bl = ld_frag2(bsl + bcol * 32 + bkk, bsl + bcol * 32 + bkk + 16);
#pragma unroll
            for (int tr = 0; tr < 2; ++tr) {
                acc[tr][ct] = wmma_bf16(af_h[tr], bh, acc[tr][ct]);
                acc[tr][ct] = wmma_bf16(af_h[tr], bl, acc[tr][ct]);
                acc[tr][ct] = wmma_bf16(af_l[tr], bh, acc[tr][ct]);
            }
        }
    }
#else
    // ---- fallback: per-thread staging, single buffer ----
    const int srow = tid >> 1;
    const int skk = (tid & 1) * 16;
    for (int kt = 0; kt < KT; ++kt) {
        const int kg = kt * 32;
        {
            const int grow = row0 + srow;
            uint4 h0 = {0, 0, 0, 0}, h1 = {0, 0, 0, 0};
            uint4 l0 = {0, 0, 0, 0}, l1 = {0, 0, 0, 0};
            if (grow < NNODE) {
                const uint4* ph = (const uint4*)(Ahi + (size_t)grow * KPAD + kg + skk);
                const uint4* pl = (const uint4*)(Alo + (size_t)grow * KPAD + kg + skk);
                h0 = ph[0]; h1 = ph[1];
                l0 = pl[0]; l1 = pl[1];
            }
            uint4* dh = (uint4*)(&As_h[0][0] + srow * 32 + skk);
            uint4* dl = (uint4*)(&As_l[0][0] + srow * 32 + skk);
            dh[0] = h0; dh[1] = h1;
            dl[0] = l0; dl[1] = l1;
        }
        {
            const int gcol = col0 + srow;
            const uint4* ph = (const uint4*)(Bhi + (size_t)gcol * KPAD + kg + skk);
            const uint4* pl = (const uint4*)(Blo + (size_t)gcol * KPAD + kg + skk);
            uint4* dh = (uint4*)(&Bs_h[0][0] + srow * 32 + skk);
            uint4* dl = (uint4*)(&Bs_l[0][0] + srow * 32 + skk);
            dh[0] = ph[0]; dh[1] = ph[1];
            dl[0] = pl[0]; dl[1] = pl[1];
        }
        __syncthreads();
        bvec16 af_h[2], af_l[2];
#pragma unroll
        for (int tr = 0; tr < 2; ++tr) {
            const int arow = wr * 32 + tr * 16 + l15;
            af_h[tr] = ld_frag2(&As_h[0][0] + arow * 32 + akk,
                                &As_h[0][0] + arow * 32 + akk + 16);
            af_l[tr] = ld_frag2(&As_l[0][0] + arow * 32 + akk,
                                &As_l[0][0] + arow * 32 + akk + 16);
        }
#pragma unroll
        for (int ct = 0; ct < 4; ++ct) {
            const int bcol = wc * 64 + ct * 16 + l15;
            bvec16 bh = ld_frag2(&Bs_h[0][0] + bcol * 32 + bkk,
                                 &Bs_h[0][0] + bcol * 32 + bkk + 16);
            bvec16 bl = ld_frag2(&Bs_l[0][0] + bcol * 32 + bkk,
                                 &Bs_l[0][0] + bcol * 32 + bkk + 16);
#pragma unroll
            for (int tr = 0; tr < 2; ++tr) {
                acc[tr][ct] = wmma_bf16(af_h[tr], bh, acc[tr][ct]);
                acc[tr][ct] = wmma_bf16(af_h[tr], bl, acc[tr][ct]);
                acc[tr][ct] = wmma_bf16(af_l[tr], bh, acc[tr][ct]);
            }
        }
        __syncthreads();
    }
#endif

    // ---- epilogue: scatter into Xg[n][b][66+i], j = b*66+i ----
#pragma unroll
    for (int tr = 0; tr < 2; ++tr) {
#pragma unroll
        for (int ct = 0; ct < 4; ++ct) {
            const int nbase = row0 + wr * 32 + tr * 16 + ((lane < 16) ? 0 : 8);
            const int j = col0 + wc * 64 + ct * 16 + l15;
            const int b = j / CIN;
            const int i = j % CIN;
#pragma unroll
            for (int r = 0; r < 8; ++r) {
                const int n = nbase + r;
                if (n < NNODE) {
                    float v = acc[tr][ct][r];
                    __bf16 h, l;
                    split_bf(v, h, l);
                    size_t off = (size_t)n * (BATCH * KKPAD) + (size_t)b * KKPAD + CIN + i;
                    Xgh[off] = h;
                    Xgl[off] = l;
                }
            }
        }
    }
}

// ---------------------------------------------------------------------------
// K3: one block per node n.
//   A  = Xg[n]  ([64 x 160] bf16 hi/lo in LDS; Xg1 for chunks 0..3, Xg2 for 4)
//   Wc = emb[n] . pool  (built on the fly per 64-column chunk, hi/lo in LDS)
// chunks: 0=f 1=i 2=o (gate), 3=g (update), 4=s (space). Same wave owns the
// same (b,h) tile across chunks, so the LSTM combine is done in registers.
// ---------------------------------------------------------------------------
__global__ __launch_bounds__(256) void k3_node(const __bf16* __restrict__ Xg1h,
                                               const __bf16* __restrict__ Xg1l,
                                               const __bf16* __restrict__ Xg2h,
                                               const __bf16* __restrict__ Xg2l,
                                               const float* __restrict__ emb,
                                               const float* __restrict__ gw,
                                               const float* __restrict__ gb,
                                               const float* __restrict__ uw,
                                               const float* __restrict__ ub,
                                               const float* __restrict__ sw,
                                               const float* __restrict__ sb,
                                               const float* __restrict__ cell,
                                               float* __restrict__ out) {
    const int n = blockIdx.x;
    const int tid = threadIdx.x;
    const int lane = tid & 31;
    const int wv = tid >> 5;
    const int l15 = lane & 15;
    const int hi16 = (lane >= 16);

    __shared__ __align__(16) __bf16 A_h[BATCH * KKPAD];   // 20 KB
    __shared__ __align__(16) __bf16 A_l[BATCH * KKPAD];   // 20 KB
    __shared__ __align__(16) __bf16 W_h[HDIM * KKPAD];    // 20 KB ([o][kk])
    __shared__ __align__(16) __bf16 W_l[HDIM * KKPAD];    // 20 KB

    float e[EDIM];
#pragma unroll
    for (int d = 0; d < EDIM; ++d) e[d] = emb[n * EDIM + d];

    const size_t nodeoff = (size_t)n * (BATCH * KKPAD);
    {
        const uint4* g1h = (const uint4*)(Xg1h + nodeoff);
        const uint4* g1l = (const uint4*)(Xg1l + nodeoff);
        for (int t = tid; t < (BATCH * KKPAD) / 8; t += 256) {
            ((uint4*)A_h)[t] = g1h[t];
            ((uint4*)A_l)[t] = g1l[t];
        }
    }

    fvec8 acc[5][2];
    const int t0 = wv * 2;
    const int akk0 = hi16 ? 8 : 0;
    const int bkk0 = hi16 ? 8 : 0;

    for (int ch = 0; ch < 5; ++ch) {
        if (ch == 4) {  // swap in A2 (previous iteration ended with a barrier)
            const uint4* g2h = (const uint4*)(Xg2h + nodeoff);
            const uint4* g2l = (const uint4*)(Xg2l + nodeoff);
            for (int t = tid; t < (BATCH * KKPAD) / 8; t += 256) {
                ((uint4*)A_h)[t] = g2h[t];
                ((uint4*)A_l)[t] = g2l[t];
            }
        }
        // ---- build W chunk: W[o][kk] = sum_d e[d] * pool[d][k][i][obase+o] ----
        const float* pool;
        int C, obase;
        if (ch < 3)       { pool = gw; C = 3 * HDIM; obase = ch * HDIM; }
        else if (ch == 3) { pool = uw; C = HDIM;     obase = 0; }
        else              { pool = sw; C = HDIM;     obase = 0; }
        {
            const int o = tid & 63;
            for (int kk = tid >> 6; kk < KKPAD; kk += 4) {
                float w = 0.0f;
                if (kk < 2 * CIN) {
                    const int k = (kk >= CIN) ? 1 : 0;
                    const int i = kk - k * CIN;
                    const int base = (k * CIN + i) * C + obase + o;
#pragma unroll
                    for (int d = 0; d < EDIM; ++d)
                        w += e[d] * pool[d * (2 * CIN * C) + base];
                }
                __bf16 h, l;
                split_bf(w, h, l);
                W_h[o * KKPAD + kk] = h;
                W_l[o * KKPAD + kk] = l;
            }
        }
        __syncthreads();

        // ---- compute this wave's 2 tiles (same (mr,nc) for every chunk) ----
#pragma unroll
        for (int tt = 0; tt < 2; ++tt) {
            const int t = t0 + tt;
            const int mr = t & 3;
            const int nc = t >> 2;
            fvec8 a = {0.f, 0.f, 0.f, 0.f, 0.f, 0.f, 0.f, 0.f};
            const int arow = mr * 16 + l15;
            const int bcol = nc * 16 + l15;
#pragma unroll
            for (int kt = 0; kt < KKPAD / 32; ++kt) {
                const int kb = kt * 32;
                bvec16 ah = ld_frag2(A_h + arow * KKPAD + kb + akk0,
                                     A_h + arow * KKPAD + kb + akk0 + 16);
                bvec16 al = ld_frag2(A_l + arow * KKPAD + kb + akk0,
                                     A_l + arow * KKPAD + kb + akk0 + 16);
                bvec16 bh = ld_frag2(W_h + bcol * KKPAD + kb + bkk0,
                                     W_h + bcol * KKPAD + kb + bkk0 + 16);
                bvec16 bl = ld_frag2(W_l + bcol * KKPAD + kb + bkk0,
                                     W_l + bcol * KKPAD + kb + bkk0 + 16);
                a = wmma_bf16(ah, bh, a);
                a = wmma_bf16(ah, bl, a);
                a = wmma_bf16(al, bh, a);
            }
            acc[ch][tt] = a;
        }
        __syncthreads();  // before next chunk overwrites W (and A at ch==4)
    }

    // ---- finale: bias + activations + LSTM combine, all in registers ----
#pragma unroll
    for (int tt = 0; tt < 2; ++tt) {
        const int t = t0 + tt;
        const int mr = t & 3;
        const int nc = t >> 2;
        const int col = nc * 16 + l15;  // h-dim column 0..63
        float bf = 0.f, bi = 0.f, bo = 0.f, bg = 0.f, bs = 0.f;
#pragma unroll
        for (int d = 0; d < EDIM; ++d) {
            bf += e[d] * gb[d * (3 * HDIM) + col];
            bi += e[d] * gb[d * (3 * HDIM) + HDIM + col];
            bo += e[d] * gb[d * (3 * HDIM) + 2 * HDIM + col];
            bg += e[d] * ub[d * HDIM + col];
            bs += e[d] * sb[d * HDIM + col];
        }
#pragma unroll
        for (int r = 0; r < 8; ++r) {
            const int b = mr * 16 + r + (hi16 ? 8 : 0);
            const float f = sigmoidf_(acc[0][tt][r] + bf);
            const float i = sigmoidf_(acc[1][tt][r] + bi);
            const float o = sigmoidf_(acc[2][tt][r] + bo);
            const float g = tanhf(acc[3][tt][r] + bg);
            const float s = sigmoidf_(acc[4][tt][r] + bs);
            const size_t idx = ((size_t)b * NNODE + n) * HDIM + col;
            const float cv = cell[idx];
            const float cnew = f * cv + i * g;
            const float h = o * tanhf(cnew);
            out[idx] = h;                  // h
            out[OUTBLK + idx] = cnew;      // c
            out[2 * OUTBLK + idx] = s;     // new_last_state
        }
    }
}

// ---------------------------------------------------------------------------
// host launcher
// ---------------------------------------------------------------------------
extern "C" void kernel_launch(void* const* d_in, const int* in_sizes, int n_in,
                              void* d_out, int out_size, void* d_ws, size_t ws_size,
                              hipStream_t stream) {
    const float* x          = (const float*)d_in[0];
    const float* y          = (const float*)d_in[1];
    const float* state      = (const float*)d_in[2];
    const float* cell       = (const float*)d_in[3];
    const float* last_state = (const float*)d_in[4];
    const float* emb        = (const float*)d_in[5];
    const float* gate_w     = (const float*)d_in[6];
    const float* gate_b     = (const float*)d_in[7];
    const float* update_w   = (const float*)d_in[8];
    const float* update_b   = (const float*)d_in[9];
    const float* space_w    = (const float*)d_in[10];
    const float* space_b    = (const float*)d_in[11];
    float* out = (float*)d_out;

    // workspace layout (bytes; all sizes multiples of 256); total ~214 MB
    char* ws = (char*)d_ws;
    const size_t SZ_SUP = (size_t)NNODE * KPAD * 2;                // 8,064,000
    const size_t SZ_XP  = (size_t)JTOT * KPAD * 2;                 // 17,031,168
    const size_t SZ_XG  = (size_t)NNODE * BATCH * KKPAD * 2;       // 40,960,000
    __bf16* sup_hi = (__bf16*)(ws);
    __bf16* sup_lo = (__bf16*)(ws + SZ_SUP);
    __bf16* Xp_hi  = (__bf16*)(ws + 2 * SZ_SUP);
    __bf16* Xp_lo  = (__bf16*)(ws + 2 * SZ_SUP + SZ_XP);
    __bf16* Xg1_hi = (__bf16*)(ws + 2 * SZ_SUP + 2 * SZ_XP);
    __bf16* Xg1_lo = (__bf16*)(ws + 2 * SZ_SUP + 2 * SZ_XP + SZ_XG);
    __bf16* Xg2_hi = (__bf16*)(ws + 2 * SZ_SUP + 2 * SZ_XP + 2 * SZ_XG);
    __bf16* Xg2_lo = (__bf16*)(ws + 2 * SZ_SUP + 2 * SZ_XP + 3 * SZ_XG);

    (void)in_sizes; (void)n_in; (void)out_size; (void)ws_size;

    // K0: supports
    k0_supports<<<NNODE, 256, 0, stream>>>(emb, sup_hi, sup_lo);

    const long T = (long)JTOT * KPAD + (long)NNODE * BATCH * (KKPAD - 2 * CIN);
    const int packBlocks = (int)((T + 255) / 256);
    dim3 gemmGrid((NNODE + 127) / 128, JTOT / 128);  // 16 x 33

    // input_and_state path (Xp buffers reused afterwards for the last-state path)
    k1_pack<<<packBlocks, 256, 0, stream>>>(x, state, Xp_hi, Xp_lo, Xg1_hi, Xg1_lo);
    k2_gemm<<<gemmGrid, 256, 0, stream>>>(sup_hi, sup_lo, Xp_hi, Xp_lo, Xg1_hi, Xg1_lo);

    // input_and_last_state path
    k1_pack<<<packBlocks, 256, 0, stream>>>(y, last_state, Xp_hi, Xp_lo, Xg2_hi, Xg2_lo);
    k2_gemm<<<gemmGrid, 256, 0, stream>>>(sup_hi, sup_lo, Xp_hi, Xp_lo, Xg2_hi, Xg2_lo);

    // per-node WMMA einsum + gating + LSTM combine
    k3_node<<<NNODE, 256, 0, stream>>>(Xg1_hi, Xg1_lo, Xg2_hi, Xg2_lo, emb,
                                       gate_w, gate_b, update_w, update_b,
                                       space_w, space_b, cell, out);
}
